// HomoGATEncoderLinearDropout_15805479649921
// MI455X (gfx1250) — matrix operand order repaired and checked by reading, not compiled, not run on hardware
//
#include <hip/hip_runtime.h>
#include <stdint.h>

typedef __attribute__((ext_vector_type(16))) _Float16 v16h;
typedef __attribute__((ext_vector_type(8)))  float    v8f;

#define NEG_SLOPE 0.2f

static __device__ __forceinline__ int f2key(float f) {
  int i = __float_as_int(f);
  return (i >= 0) ? i : (i ^ 0x7FFFFFFF);
}
static __device__ __forceinline__ float key2f(int k) {
  int i = (k >= 0) ? k : (k ^ 0x7FFFFFFF);
  return __int_as_float(i);
}
static __device__ __forceinline__ float elu_f(float v) {
  return v > 0.f ? v : (__expf(v) - 1.f);
}

// ---------------- conversion / init kernels ----------------

__global__ void k_f32_to_f16(const float* __restrict__ src, _Float16* __restrict__ dst, long long n) {
  long long i = (long long)blockIdx.x * blockDim.x + threadIdx.x;
  if (i < n) dst[i] = (_Float16)src[i];
}

// W [K,Nc] row-major f32  ->  Wt [Nc,K] row-major f16
__global__ void k_transpose_f32_to_f16(const float* __restrict__ W, _Float16* __restrict__ Wt,
                                       int K, int Nc) {
  int i = blockIdx.x * blockDim.x + threadIdx.x;
  if (i < K * Nc) {
    int k = i / Nc, n = i % Nc;
    Wt[(long long)n * K + k] = (_Float16)W[i];
  }
}

__global__ void k_fill_f32(float* __restrict__ p, float v, long long n) {
  long long i = (long long)blockIdx.x * blockDim.x + threadIdx.x;
  if (i < n) p[i] = v;
}

__global__ void k_fill_i32(int* __restrict__ p, int v, long long n) {
  long long i = (long long)blockIdx.x * blockDim.x + threadIdx.x;
  if (i < n) p[i] = v;
}

__global__ void k_init_bias(float* __restrict__ out, const float* __restrict__ bias,
                            int rowlen, long long total) {
  long long i = (long long)blockIdx.x * blockDim.x + threadIdx.x;
  if (i < total) out[i] = bias[(int)(i % rowlen)];
}

// ---------------- WMMA GEMM: C[M,Nc] = A[M,K](f16) @ Bt[Nc,K]^T(f16) + bias ----------------
// One wave -> one 16x16 tile; K stepped by 32 via v_wmma_f32_16x16x32_f16.
// A-fragment per documented 16-bit A layout: lane (m = lane&15), VGPR pair p holds
//   K = (p<4 ? 2p : 16+2(p-4)) + 8*(lane>>4)  and K+1.
// B-fragment (column-major weights pre-transposed): lane (n = lane&15), pair p holds
//   K = (p<4 ? 2p : 16+2(p-4)) + 8*(lane>>4) and K+1.
__global__ void k_wmma_gemm_bias(const _Float16* __restrict__ A,
                                 const _Float16* __restrict__ Bt,
                                 const float* __restrict__ bias,
                                 float* __restrict__ C,
                                 int M, int K, int Nc) {
  const int lane = threadIdx.x & 31;
  const int wave = threadIdx.x >> 5;
  const int tiles_n = Nc >> 4;
  const int tiles_m = (M + 15) >> 4;
  const int tile = blockIdx.x * (blockDim.x >> 5) + wave;   // wave-uniform
  if (tile >= tiles_m * tiles_n) return;                    // whole-wave exit, EXEC stays full

  const int tm = tile / tiles_n;
  const int tn = tile % tiles_n;
  const int hi = lane >> 4;     // 0 | 1
  const int r  = lane & 15;

  int arow = tm * 16 + r;
  if (arow >= M) arow = M - 1;                              // clamp loads, guard stores
  const _Float16* __restrict__ Arow = A  + (long long)arow * K;
  const _Float16* __restrict__ Brow = Bt + (long long)(tn * 16 + r) * K;

  union U { v16h v; unsigned u[8]; };
  v8f acc = {};

  for (int k0 = 0; k0 < K; k0 += 32) {
    U a, b;
#pragma unroll
    for (int p = 0; p < 8; ++p) {
      const int ka = ((p < 4) ? (2 * p) : (16 + 2 * (p - 4))) + 8 * hi;
      a.u[p] = *(const unsigned*)(Arow + k0 + ka);
      const int kb = ((p < 4) ? (2 * p) : (16 + 2 * (p - 4))) + 8 * hi;
      b.u[p] = *(const unsigned*)(Brow + k0 + kb);
    }
    acc = __builtin_amdgcn_wmma_f32_16x16x32_f16(
        /*neg_a=*/false, a.v, /*neg_b=*/false, b.v,
        /*c_mod=*/(short)0, acc, /*reuse_a=*/false, /*reuse_b=*/false);
  }

  const int ncol = tn * 16 + r;
  const float bv = bias ? bias[ncol] : 0.f;
#pragma unroll
  for (int vr = 0; vr < 8; ++vr) {
    const int grow = tm * 16 + vr + 8 * hi;   // C/D layout: VGPR vr -> M = vr (+8 upper half)
    if (grow < M) C[(long long)grow * Nc + ncol] = acc[vr] + bv;
  }
}

// ---------------- GATv2 edge kernels (self-loops appended: e >= E -> src=dst=e-E) ----------

__global__ void k_edge_logits(const long long* __restrict__ ei, int E, int N, int H, int C,
                              const float* __restrict__ xl, const float* __restrict__ xr,
                              const float* __restrict__ att,
                              float* __restrict__ logits, int* __restrict__ keys) {
  const long long tot = ((long long)E + N) * H;
  long long gid = (long long)blockIdx.x * blockDim.x + threadIdx.x;
  if (gid >= tot) return;
  const int h = (int)(gid % H);
  const long long e = gid / H;
  long long src, dst;
  if (e < E) { src = ei[e]; dst = ei[(long long)E + e]; } else { src = dst = e - (long long)E; }

  const float* __restrict__ pl = xl + (src * H + h) * (long long)C;
  const float* __restrict__ pr = xr + (dst * H + h) * (long long)C;
  const float* __restrict__ pa = att + (long long)h * C;
  __builtin_prefetch(pl, 0, 1);   // global_prefetch_b8
  __builtin_prefetch(pr, 0, 1);

  float s = 0.f;
  for (int c = 0; c < C; c += 4) {
    const float4 vl = *(const float4*)(pl + c);
    const float4 vr = *(const float4*)(pr + c);
    const float4 va = *(const float4*)(pa + c);
    float t;
    t = vl.x + vr.x; s += va.x * (t > 0.f ? t : NEG_SLOPE * t);
    t = vl.y + vr.y; s += va.y * (t > 0.f ? t : NEG_SLOPE * t);
    t = vl.z + vr.z; s += va.z * (t > 0.f ? t : NEG_SLOPE * t);
    t = vl.w + vr.w; s += va.w * (t > 0.f ? t : NEG_SLOPE * t);
  }
  logits[e * H + h] = s;
  atomicMax(keys + dst * H + h, f2key(s));  // segment max (order-preserving int key)
}

__global__ void k_edge_softmax(float* __restrict__ logits, const long long* __restrict__ ei,
                               int E, int N, int H,
                               const int* __restrict__ keys, float* __restrict__ denom) {
  const long long tot = ((long long)E + N) * H;
  long long gid = (long long)blockIdx.x * blockDim.x + threadIdx.x;
  if (gid >= tot) return;
  const int h = (int)(gid % H);
  const long long e = gid / H;
  const long long dst = (e < E) ? ei[(long long)E + e] : e - (long long)E;
  float m = key2f(keys[dst * H + h]);
  if (!isfinite(m)) m = 0.f;                      // reference parity for empty segments
  const float p = __expf(logits[e * H + h] - m);
  logits[e * H + h] = p;                          // overwrite with numerator
  atomicAdd(denom + dst * H + h, p);
}

__global__ void k_edge_aggregate(const float* __restrict__ pbuf, const long long* __restrict__ ei,
                                 int E, int N, int H, int C,
                                 const float* __restrict__ denom,
                                 const float* __restrict__ xl, float* __restrict__ out) {
  const long long tot = ((long long)E + N) * H;
  long long gid = (long long)blockIdx.x * blockDim.x + threadIdx.x;
  if (gid >= tot) return;
  const int h = (int)(gid % H);
  const long long e = gid / H;
  long long src, dst;
  if (e < E) { src = ei[e]; dst = ei[(long long)E + e]; } else { src = dst = e - (long long)E; }

  const float d = denom[dst * H + h];
  const float alpha = pbuf[e * H + h] / fmaxf(d, 1e-16f);
  const float* __restrict__ pl = xl + (src * H + h) * (long long)C;
  float* __restrict__ po = out + (dst * H + h) * (long long)C;
  for (int c = 0; c < C; c += 4) {
    const float4 v = *(const float4*)(pl + c);
    atomicAdd(po + c + 0, alpha * v.x);
    atomicAdd(po + c + 1, alpha * v.y);
    atomicAdd(po + c + 2, alpha * v.z);
    atomicAdd(po + c + 3, alpha * v.w);
  }
}

__global__ void k_elu_to_f16(const float* __restrict__ in, _Float16* __restrict__ outh, long long n) {
  long long i = (long long)blockIdx.x * blockDim.x + threadIdx.x;
  if (i < n) outh[i] = (_Float16)elu_f(in[i]);
}

// out[N,16] = elu(h2[N,32]) @ Wlin[32,16] + blin
__global__ void k_final_linear(const float* __restrict__ h2, const float* __restrict__ Wlin,
                               const float* __restrict__ blin, float* __restrict__ out, int N) {
  int gid = blockIdx.x * blockDim.x + threadIdx.x;
  if (gid >= N * 16) return;
  const int n = gid >> 4, o = gid & 15;
  const float* __restrict__ ph = h2 + (long long)n * 32;
  float acc = blin[o];
#pragma unroll
  for (int c = 0; c < 32; ++c) acc += elu_f(ph[c]) * Wlin[c * 16 + o];
  out[gid] = acc;
}

// ---------------- launch ----------------

static inline unsigned nblk(long long n, int t) { return (unsigned)((n + t - 1) / t); }

extern "C" void kernel_launch(void* const* d_in, const int* in_sizes, int n_in,
                              void* d_out, int out_size, void* d_ws, size_t ws_size,
                              hipStream_t stream) {
  const float*     x    = (const float*)d_in[0];
  const long long* ei   = (const long long*)d_in[1];
  const float*     W1l  = (const float*)d_in[2];
  const float*     W1r  = (const float*)d_in[3];
  const float*     att1 = (const float*)d_in[4];
  const float*     b1   = (const float*)d_in[5];
  const float*     W2l  = (const float*)d_in[6];
  const float*     W2r  = (const float*)d_in[7];
  const float*     att2 = (const float*)d_in[8];
  const float*     b2   = (const float*)d_in[9];
  const float*     Wlin = (const float*)d_in[10];
  const float*     blin = (const float*)d_in[11];
  float* out = (float*)d_out;

  const int DIN = 128, H1 = 8, C = 32, HC1 = 256;
  const int N = in_sizes[0] / DIN;
  const int E = in_sizes[1] / 2;
  const long long Etot = (long long)E + N;

  // ---- workspace layout (aliased; ~185 MB peak) ----
  char* ws = (char*)d_ws;
  size_t off = 0;
  auto alloc = [&](size_t bytes) -> char* {
    char* p = ws + off;
    off = (off + bytes + 255) & ~(size_t)255;
    return p;
  };
  _Float16* xh    = (_Float16*)alloc((size_t)N * DIN * 2);
  _Float16* W1lt  = (_Float16*)alloc((size_t)HC1 * DIN * 2);
  _Float16* W1rt  = (_Float16*)alloc((size_t)HC1 * DIN * 2);
  _Float16* W2lt  = (_Float16*)alloc((size_t)C * HC1 * 2);
  _Float16* W2rt  = (_Float16*)alloc((size_t)C * HC1 * 2);
  float* xl1    = (float*)alloc((size_t)N * HC1 * 4);   // 51.2 MB (L2-resident)
  float* xr1    = (float*)alloc((size_t)N * HC1 * 4);   // 51.2 MB (L2-resident)
  float* logits = (float*)alloc((size_t)Etot * H1 * 4);
  int*   keys   = (int*)alloc((size_t)N * H1 * 4);
  float* denom  = (float*)alloc((size_t)N * H1 * 4);
  float* out1   = (float*)alloc((size_t)N * HC1 * 4);
  // aliases (dead-buffer reuse after layer-1 aggregation):
  _Float16* hh  = (_Float16*)xl1;                 // [N,256] f16 layer-2 input
  float* xl2    = xr1;                            // [N,32]
  float* xr2    = xr1 + (size_t)N * C;            // [N,32]
  float* log2b  = logits;                         // [Etot,1]
  int*   keys2  = keys;
  float* den2   = denom;
  float* out2   = out1;                           // [N,32]

  const int T = 256;

  // ---- f16 copies (WMMA operand precision) ----
  k_f32_to_f16<<<nblk((long long)N * DIN, T), T, 0, stream>>>(x, xh, (long long)N * DIN);
  k_transpose_f32_to_f16<<<nblk(DIN * HC1, T), T, 0, stream>>>(W1l, W1lt, DIN, HC1);
  k_transpose_f32_to_f16<<<nblk(DIN * HC1, T), T, 0, stream>>>(W1r, W1rt, DIN, HC1);
  k_transpose_f32_to_f16<<<nblk(HC1 * C, T), T, 0, stream>>>(W2l, W2lt, HC1, C);
  k_transpose_f32_to_f16<<<nblk(HC1 * C, T), T, 0, stream>>>(W2r, W2rt, HC1, C);

  // ---- layer 1: xl1 = x @ W1l, xr1 = x @ W1r  (WMMA, f32 accumulate) ----
  {
    const int tiles = ((N + 15) / 16) * (HC1 / 16);
    k_wmma_gemm_bias<<<nblk(tiles, 4), 128, 0, stream>>>(xh, W1lt, nullptr, xl1, N, DIN, HC1);
    k_wmma_gemm_bias<<<nblk(tiles, 4), 128, 0, stream>>>(xh, W1rt, nullptr, xr1, N, DIN, HC1);
  }

  // ---- layer 1: GATv2 edge softmax + aggregation ----
  k_fill_i32<<<nblk((long long)N * H1, T), T, 0, stream>>>(keys, (int)0x80000000, (long long)N * H1);
  k_fill_f32<<<nblk((long long)N * H1, T), T, 0, stream>>>(denom, 0.f, (long long)N * H1);
  k_edge_logits<<<nblk(Etot * H1, T), T, 0, stream>>>(ei, E, N, H1, C, xl1, xr1, att1, logits, keys);
  k_edge_softmax<<<nblk(Etot * H1, T), T, 0, stream>>>(logits, ei, E, N, H1, keys, denom);
  k_init_bias<<<nblk((long long)N * HC1, T), T, 0, stream>>>(out1, b1, HC1, (long long)N * HC1);
  k_edge_aggregate<<<nblk(Etot * H1, T), T, 0, stream>>>(logits, ei, E, N, H1, C, denom, xl1, out1);

  // ---- ELU -> f16 layer-2 input (overwrites dead xl1 region) ----
  k_elu_to_f16<<<nblk((long long)N * HC1, T), T, 0, stream>>>(out1, hh, (long long)N * HC1);

  // ---- layer 2: xl2 = h @ W2l, xr2 = h @ W2r (WMMA) ----
  {
    const int tiles = ((N + 15) / 16) * (C / 16);
    k_wmma_gemm_bias<<<nblk(tiles, 4), 128, 0, stream>>>(hh, W2lt, nullptr, xl2, N, HC1, C);
    k_wmma_gemm_bias<<<nblk(tiles, 4), 128, 0, stream>>>(hh, W2rt, nullptr, xr2, N, HC1, C);
  }

  // ---- layer 2: GATv2 (heads = 1) ----
  k_fill_i32<<<nblk(N, T), T, 0, stream>>>(keys2, (int)0x80000000, (long long)N);
  k_fill_f32<<<nblk(N, T), T, 0, stream>>>(den2, 0.f, (long long)N);
  k_edge_logits<<<nblk(Etot, T), T, 0, stream>>>(ei, E, N, 1, C, xl2, xr2, att2, log2b, keys2);
  k_edge_softmax<<<nblk(Etot, T), T, 0, stream>>>(log2b, ei, E, N, 1, keys2, den2);
  k_init_bias<<<nblk((long long)N * C, T), T, 0, stream>>>(out2, b2, C, (long long)N * C);
  k_edge_aggregate<<<nblk(Etot, T), T, 0, stream>>>(log2b, ei, E, N, 1, C, den2, xl2, out2);

  // ---- elu + final linear ----
  k_final_linear<<<nblk((long long)N * 16, T), T, 0, stream>>>(out2, Wlin, blin, out, N);
}
